// MultiHeadAttention_59519656788388
// MI455X (gfx1250) — matrix operand order repaired and checked by reading, not compiled, not run on hardware
//
#include <hip/hip_runtime.h>

// MI455X / gfx1250 implementation of the double-softmax 4D attention.
// B=2, L=64, F=32, C=64, H=8.  R = L*F = 2048 rows per (b), N = 2048 key positions.
// Compute-bound (~36 GFLOP vs ~25MB traffic @ 23.3TB/s) -> f16 WMMA everywhere.

#define BB 2
#define LL 64
#define FF 32
#define CC 64
#define HH 8
#define RR 2048          // L*F
#define NPAD 2056        // 2048 + 8 f16 pad -> row stride 4112B, banks rotate by 4

typedef _Float16 v8h  __attribute__((ext_vector_type(8)));
typedef _Float16 v16h __attribute__((ext_vector_type(16)));
typedef float    v8f  __attribute__((ext_vector_type(8)));

__device__ __forceinline__ v16h cat8(v8h lo, v8h hi) {
    return __builtin_shufflevector(lo, hi, 0,1,2,3,4,5,6,7,8,9,10,11,12,13,14,15);
}

// A-operand 16x32 f16: lane = row (mod 16); half-wave K permutation per ISA:
// lanes 0-15 hold K {kb+0..7, kb+16..23}; lanes 16-31 hold K {kb+8..15, kb+24..31}.
__device__ __forceinline__ v16h ldA(const _Float16* row, int kb, int hf) {
    v8h lo = *(const v8h*)(row + kb + hf * 8);
    v8h hi = *(const v8h*)(row + kb + 16 + hf * 8);
    return cat8(lo, hi);
}
// Same, sourcing fp32 and converting.
__device__ __forceinline__ v16h ldA_f32(const float* row, int kb, int hf) {
    v16h a;
    const float* p = row + kb + hf * 8;
#pragma unroll
    for (int j = 0; j < 8; ++j) { a[j] = (_Float16)p[j]; a[8 + j] = (_Float16)p[16 + j]; }
    return a;
}
// B-operand 32x16 f16 from a "Bt" matrix stored as rows of K-contiguous data
// (Bt[n][k] = B[k][n]): lane = column n (mod 16); lanes 0-15 K {kb+0..7, kb+16..23},
// lanes 16-31 K {kb+8..15, kb+24..31} (same K permutation as A).
__device__ __forceinline__ v16h ldB(const _Float16* rowN, int kb, int hf) {
    const _Float16* p = rowN + kb + hf * 8;
    return cat8(*(const v8h*)p, *(const v8h*)(p + 16));
}

__device__ __forceinline__ v8f wmma_f16(v16h a, v16h b, v8f c) {
    return __builtin_amdgcn_wmma_f32_16x16x32_f16(false, a, false, b, (short)0, c, false, false);
}

// ---------------------------------------------------------------------------
// Kernel 1: repack weights to f16 in WMMA-B-friendly layouts.
//   Wt[p][h][c][cc]  = W_p[cc, c*H+h]   (p=0 scaled by 1/sqrt(C)=0.125)
//   bhd[p][h][c]     = b_p[c*H+h]       (p=0 scaled by 0.125)
//   WoT[c][k]        = Wo[k, c]
// ---------------------------------------------------------------------------
__global__ void prep_kernel(const float* Wq, const float* bq, const float* Wk, const float* bk,
                            const float* Wv, const float* bv, const float* Wo,
                            _Float16* Wt, _Float16* bhd, _Float16* WoT) {
    int total = 3 * 32768 + 3 * 512 + 32768;
    for (int i = blockIdx.x * blockDim.x + threadIdx.x; i < total; i += gridDim.x * blockDim.x) {
        if (i < 3 * 32768) {
            int p = i / 32768, j = i % 32768;
            int h = j / 4096, rem = j % 4096, c = rem / 64, cc = rem % 64;
            const float* W = (p == 0) ? Wq : (p == 1) ? Wk : Wv;
            float s = (p == 0) ? 0.125f : 1.0f;
            Wt[i] = (_Float16)(W[cc * (CC * HH) + c * HH + h] * s);
        } else if (i < 3 * 32768 + 3 * 512) {
            int j = i - 3 * 32768;
            int p = j / 512, k2 = j % 512, h = k2 / 64, c = k2 % 64;
            const float* bb = (p == 0) ? bq : (p == 1) ? bk : bv;
            float s = (p == 0) ? 0.125f : 1.0f;
            bhd[j] = (_Float16)(bb[c * HH + h] * s);
        } else {
            int j = i - 3 * 32768 - 1536;
            int c = j / 512, k2 = j % 512;
            WoT[j] = (_Float16)(Wo[k2 * CC + c]);
        }
    }
}

// ---------------------------------------------------------------------------
// Kernel 2: Q/K/V projections via WMMA.  One wave per 16x16 output tile.
//   qws,kws: per (b,h) row-major (2048 x 64) f16 ;  vtws: transposed (64 x 2048) f16
// ---------------------------------------------------------------------------
__global__ __launch_bounds__(256) void proj_kernel(const float* qin, const float* kin, const float* vin,
                                                   const _Float16* Wt, const _Float16* bhd,
                                                   _Float16* qws, _Float16* kws, _Float16* vtws) {
    int tid = threadIdx.x, wv = tid >> 5, lane = tid & 31, l16 = lane & 15, hf = lane >> 4;
    int t = blockIdx.x * 8 + wv;             // 0..8191
    int ct = t & 3, rt = (t >> 2) & 127, bh = t >> 9;
    int b = bh >> 3, h = bh & 7;
    int row0 = rt * 16, c0 = ct * 16, c = c0 + l16;
    const float* ins[3] = {qin, kin, vin};
#pragma unroll
    for (int p = 0; p < 3; ++p) {
        const float* xrow = ins[p] + (size_t)(b * RR + row0 + l16) * CC;
        const _Float16* wrow = Wt + ((size_t)p * 8 + h) * 4096 + (size_t)c * 64;
        v8f acc = {};
        acc = wmma_f16(ldA_f32(xrow, 0, hf), ldB(wrow, 0, hf), acc);
        acc = wmma_f16(ldA_f32(xrow, 32, hf), ldB(wrow, 32, hf), acc);
        float bias = (float)bhd[p * 512 + h * 64 + c];
        if (p < 2) {
            _Float16* dst = (p == 0 ? qws : kws) + (size_t)bh * RR * CC;
#pragma unroll
            for (int i = 0; i < 8; ++i) {
                int r = row0 + i + 8 * hf;
                dst[(size_t)r * CC + c] = (_Float16)(acc[i] + bias);
            }
        } else {
            _Float16* dst = vtws + (size_t)bh * CC * RR;
#pragma unroll
            for (int i = 0; i < 8; ++i) {
                int r = row0 + i + 8 * hf;
                dst[(size_t)c * RR + r] = (_Float16)(acc[i] + bias);
            }
        }
    }
}

// ---------------------------------------------------------------------------
// Kernel 3: fused scores + double softmax + weighted sum of V.
// One 256-thread workgroup per (bh, 16-query-row block). Both softmaxes are
// row-local over n (n = m*32+g), so the 16x2048 block never leaves LDS.
// ---------------------------------------------------------------------------
__global__ __launch_bounds__(256) void attn_kernel(const _Float16* qws, const _Float16* kws,
                                                   const _Float16* vtws, float* zcat) {
    __shared__ _Float16 sExp[16][NPAD];   // exp tiles (f16), ~66KB
    __shared__ float sCol[32][16];        // [g][r] column-class sums
    __shared__ float sRow[64][16];        // [m][r] row-block sums
    __shared__ float sZ[16][64];          // output reduction

    int tid = threadIdx.x, wv = tid >> 5, lane = tid & 31, l16 = lane & 15, hf = lane >> 4;
    int bh = blockIdx.x >> 7, rb = blockIdx.x & 127;
    int b = bh >> 3, h = bh & 7;
    int row0 = rb * 16;

    for (int i = tid; i < 512; i += 256) ((float*)sCol)[i] = 0.f;
    for (int i = tid; i < 1024; i += 256) ((float*)sRow)[i] = 0.f;
    for (int i = tid; i < 1024; i += 256) ((float*)sZ)[i] = 0.f;
    __syncthreads();

    // Q fragments for this 16-row block (scale 1/8 already folded into Wq)
    const _Float16* qrow = qws + ((size_t)bh * RR + row0 + l16) * CC;
    v16h a0 = ldA(qrow, 0, hf), a1 = ldA(qrow, 32, hf);

    // ---- pass 1: S tiles -> exp(S) into LDS + per-(g,r) column sums ----
    for (int nt = wv * 16; nt < wv * 16 + 16; ++nt) {
        const _Float16* krow = kws + ((size_t)bh * RR + nt * 16 + l16) * CC;
        v8f d = {};
        d = wmma_f16(a0, ldB(krow, 0, hf), d);
        d = wmma_f16(a1, ldB(krow, 32, hf), d);
        int n = nt * 16 + l16, g = n & 31;
#pragma unroll
        for (int i = 0; i < 8; ++i) {
            float e = __expf(d[i]);          // |s| << 1, no max-shift needed
            int r = i + 8 * hf;
            sExp[r][n] = (_Float16)e;
            atomicAdd(&sCol[g][r], e);       // ds_add_f32
        }
    }
    __syncthreads();
    for (int i = tid; i < 512; i += 256) ((float*)sCol)[i] = 1.0f / ((float*)sCol)[i];
    __syncthreads();

    // ---- pass 2: w1 = exp(s)/colsum ; store exp(w1), per-(m,r) row sums ----
    {
        int r = tid & 15, cb = tid >> 4;
        for (int nb = cb; nb < 128; nb += 16) {
            int n0 = nb * 16, m = n0 >> 5;
            float local = 0.f;
#pragma unroll
            for (int j = 0; j < 16; ++j) {
                int n = n0 + j;
                float w1 = (float)sExp[r][n] * sCol[n & 31][r];
                float e2 = __expf(w1);       // w1 in (0,1]
                sExp[r][n] = (_Float16)e2;
                local += e2;
            }
            atomicAdd(&sRow[m][r], local);
        }
    }
    __syncthreads();
    for (int i = tid; i < 1024; i += 256) ((float*)sRow)[i] = 1.0f / ((float*)sRow)[i];
    __syncthreads();

    // ---- pass 3: Z = W2 @ V.  1/rowsum[r][m] is constant over each 16x32
    // A-fragment (K-block == one m), so fold it into the A load. ----
    {
        v8f acc0 = {}, acc1 = {}, acc2 = {}, acc3 = {};
        const _Float16* vtb = vtws + (size_t)bh * CC * RR;
        for (int kt = 0; kt < 8; ++kt) {
            int kb = wv * 256 + kt * 32;
            int m = kb >> 5;
            _Float16 sc = (_Float16)sRow[m][l16];
            v8h lo = *(const v8h*)&sExp[l16][kb + hf * 8];
            v8h hi = *(const v8h*)&sExp[l16][kb + 16 + hf * 8];
            lo = lo * sc; hi = hi * sc;
            v16h a = cat8(lo, hi);
            acc0 = wmma_f16(a, ldB(vtb + (size_t)(0  + l16) * RR, kb, hf), acc0);
            acc1 = wmma_f16(a, ldB(vtb + (size_t)(16 + l16) * RR, kb, hf), acc1);
            acc2 = wmma_f16(a, ldB(vtb + (size_t)(32 + l16) * RR, kb, hf), acc2);
            acc3 = wmma_f16(a, ldB(vtb + (size_t)(48 + l16) * RR, kb, hf), acc3);
        }
#pragma unroll
        for (int i = 0; i < 8; ++i) {
            int r = i + 8 * hf;
            atomicAdd(&sZ[r][ 0 + l16], acc0[i]);
            atomicAdd(&sZ[r][16 + l16], acc1[i]);
            atomicAdd(&sZ[r][32 + l16], acc2[i]);
            atomicAdd(&sZ[r][48 + l16], acc3[i]);
        }
    }
    __syncthreads();

    // scatter into head-fastest concat layout: zcat[b*R + row][c*H + h]
    for (int i = tid; i < 1024; i += 256) {
        int r = i >> 6, c = i & 63;
        zcat[((size_t)(b * RR) + row0 + r) * (CC * HH) + c * HH + h] = sZ[r][c];
    }
}

// ---------------------------------------------------------------------------
// Kernel 4: out = zcat(4096x512) @ Wo(512x64) + bo, via WMMA (K=512).
// ---------------------------------------------------------------------------
__global__ __launch_bounds__(256) void out_kernel(const float* zcat, const _Float16* WoT,
                                                  const float* bo, float* out) {
    int tid = threadIdx.x, wv = tid >> 5, lane = tid & 31, l16 = lane & 15, hf = lane >> 4;
    int t = blockIdx.x * 8 + wv;   // 0..1023
    int ct = t & 3, rt = t >> 2;   // 256 row tiles x 4 col tiles
    int row0 = rt * 16, c0 = ct * 16, c = c0 + l16;
    const float* zrow = zcat + (size_t)(row0 + l16) * (CC * HH);
    const _Float16* wrow = WoT + (size_t)c * (CC * HH);
    v8f acc = {};
#pragma unroll
    for (int k = 0; k < 16; ++k)
        acc = wmma_f16(ldA_f32(zrow, k * 32, hf), ldB(wrow, k * 32, hf), acc);
    float bias = bo[c];
#pragma unroll
    for (int i = 0; i < 8; ++i) {
        int r = row0 + i + 8 * hf;
        out[(size_t)r * CC + c] = acc[i] + bias;
    }
}

// ---------------------------------------------------------------------------
extern "C" void kernel_launch(void* const* d_in, const int* in_sizes, int n_in,
                              void* d_out, int out_size, void* d_ws, size_t ws_size,
                              hipStream_t stream) {
    (void)in_sizes; (void)n_in; (void)out_size; (void)ws_size;
    const float* query = (const float*)d_in[0];
    const float* key   = (const float*)d_in[1];
    const float* value = (const float*)d_in[2];
    const float* Wq = (const float*)d_in[3];
    const float* bq = (const float*)d_in[4];
    const float* Wk = (const float*)d_in[5];
    const float* bk = (const float*)d_in[6];
    const float* Wv = (const float*)d_in[7];
    const float* bv = (const float*)d_in[8];
    const float* Wo = (const float*)d_in[9];
    const float* bo = (const float*)d_in[10];

    char* ws = (char*)d_ws;
    size_t off = 0;
    _Float16* qws  = (_Float16*)(ws + off); off += (size_t)16 * RR * CC * 2;   // 4 MB
    _Float16* kws  = (_Float16*)(ws + off); off += (size_t)16 * RR * CC * 2;   // 4 MB
    _Float16* vtws = (_Float16*)(ws + off); off += (size_t)16 * CC * RR * 2;   // 4 MB
    float*    zcat = (float*)(ws + off);    off += (size_t)(BB * RR) * (CC * HH) * 4; // 8 MB
    _Float16* Wt   = (_Float16*)(ws + off); off += (size_t)3 * 8 * 64 * 64 * 2;
    _Float16* bhd  = (_Float16*)(ws + off); off += (size_t)3 * 512 * 2;
    _Float16* WoT  = (_Float16*)(ws + off); off += (size_t)64 * 512 * 2;

    prep_kernel<<<dim3(64), dim3(256), 0, stream>>>(Wq, bq, Wk, bk, Wv, bv, Wo, Wt, bhd, WoT);
    proj_kernel<<<dim3(1024), dim3(256), 0, stream>>>(query, key, value, Wt, bhd, qws, kws, vtws);
    attn_kernel<<<dim3(2048), dim3(256), 0, stream>>>(qws, kws, vtws, zcat);
    out_kernel<<<dim3(128), dim3(256), 0, stream>>>(zcat, WoT, bo, (float*)d_out);
}
